// mrg_56281251447038
// MI455X (gfx1250) — matrix operand order repaired and checked by reading, not compiled, not run on hardware
//
#include <hip/hip_runtime.h>
#include <hip/hip_bf16.h>
#include <math.h>

// ---- model dims ----
#define MF_      64
#define MLPO_    64
#define H_       512
#define WD_      512
#define V_       20000
#define B_       128
#define T_       40
#define AVG_     3.5f
#define GI_      (3 * H_)        // 1536
#define XD_      (WD_ + MLPO_)   // 576

typedef __attribute__((ext_vector_type(16))) _Float16 v16h;
typedef __attribute__((ext_vector_type(8)))  _Float16 v8h;
typedef __attribute__((ext_vector_type(8)))  float    v8f;

// -------------------------------------------------------------------------
// WMMA fragment loaders (contiguous k order, same in A and B, wave32)
// A: [M,K] row-major f16. Lane l holds row (l&15); v16h elems 0..15 = K base..base+15,
//    base = (l>>4)*16. Two contiguous 16B loads.
// B: weights stored [N,K] row-major; WMMA B is Kx16 col n. Lane l -> col n0+(l&15),
//    K run (l>>4)*16 .. +15: one contiguous 32B load.
// -------------------------------------------------------------------------
__device__ __forceinline__ v16h load_afrag(const _Float16* __restrict__ A, int lda,
                                           int m0, int k0, int lane) {
  const int lr = lane & 15, hi = lane >> 4;
  const _Float16* p = A + (size_t)(m0 + lr) * lda + k0 + hi * 16;
  v8h a0 = *(const v8h*)(p);
  v8h a1 = *(const v8h*)(p + 8);
  v16h a;
#pragma unroll
  for (int j = 0; j < 8; ++j) { a[j] = a0[j]; a[j + 8] = a1[j]; }
  return a;
}

__device__ __forceinline__ v16h load_bfrag(const _Float16* __restrict__ W, int ldb,
                                           int n0, int k0, int lane) {
  const int ln = lane & 15, kh = lane >> 4;
  return *(const v16h*)(W + (size_t)(n0 + ln) * ldb + k0 + kh * 16);
}

// C[M,N] = A[M,K] @ W[N,K]^T + bias[N]   (f16 in, f32 accumulate/out)
// One wave computes a (16*MACC) x (16*NACC) macro-tile: MACC*NACC WMMAs per K-step
// against 2*(MACC+NACC) b128 loads. Sized to avoid VGPR spills (acc <= 64 VGPRs).
// Requires M % (16*MACC) == 0, N % (16*NACC) == 0, K % 32 == 0.
template <int MACC, int NACC>
__global__ void gemm_wmma_kernel(const _Float16* __restrict__ A,
                                 const _Float16* __restrict__ W,
                                 const float* __restrict__ bias,
                                 float* __restrict__ C, int M, int N, int K) {
  const int lane = threadIdx.x & 31;
  const int wid  = blockIdx.x * (blockDim.x >> 5) + (threadIdx.x >> 5);
  const int Ntg  = N / (16 * NACC);
  const int mg   = wid / Ntg;
  const int ntg  = wid % Ntg;
  if (mg * 16 * MACC >= M) return;   // wave-uniform: EXEC stays all-ones for WMMA

  v8f acc[MACC][NACC];
#pragma unroll
  for (int mi = 0; mi < MACC; ++mi)
#pragma unroll
    for (int a = 0; a < NACC; ++a)
#pragma unroll
      for (int r = 0; r < 8; ++r) acc[mi][a][r] = 0.0f;

  const int m0 = mg * 16 * MACC;
  const int n0 = ntg * 16 * NACC;

  for (int k0 = 0; k0 < K; k0 += 32) {
    v16h af[MACC];
#pragma unroll
    for (int mi = 0; mi < MACC; ++mi)
      af[mi] = load_afrag(A, K, m0 + mi * 16, k0, lane);
#pragma unroll
    for (int a = 0; a < NACC; ++a) {
      const v16h bf = load_bfrag(W, K, n0 + a * 16, k0, lane);
#pragma unroll
      for (int mi = 0; mi < MACC; ++mi)
        acc[mi][a] = __builtin_amdgcn_wmma_f32_16x16x32_f16(
            false, af[mi], false, bf, (short)0, acc[mi][a], false, false);
    }
  }

  // C/D layout: lane l -> col n0+(l&15); VGPR r -> row base + (l>>4)*8 + r
  const int colb = n0 + (lane & 15);
  const int rowb = m0 + (lane >> 4) * 8;
#pragma unroll
  for (int a = 0; a < NACC; ++a) {
    const int col = colb + a * 16;
    const float bv = bias ? bias[col] : 0.0f;
#pragma unroll
    for (int mi = 0; mi < MACC; ++mi)
#pragma unroll
      for (int r = 0; r < 8; ++r)
        C[(size_t)(rowb + mi * 16 + r) * N + col] = acc[mi][a][r] + bv;
  }
}

// -------------------------------------------------------------------------
__global__ void cvt_f16_kernel(const float* __restrict__ src,
                               _Float16* __restrict__ dst, int n) {
  int i = blockIdx.x * blockDim.x + threadIdx.x;
  if (i < n) dst[i] = (_Float16)src[i];
}

__global__ void init_h_kernel(float* __restrict__ h, _Float16* __restrict__ h_h) {
  int i = blockIdx.x * blockDim.x + threadIdx.x;
  if (i < B_ * H_) { h[i] = 0.0f; h_h[i] = (_Float16)0.0f; }
}

// Encoder + rating head: m = tanh([ue,ie] @ mlp_W^T); pred -> rating_out_loss
__global__ void encoder_kernel(const float* __restrict__ user_emb,
                               const float* __restrict__ item_emb,
                               const float* __restrict__ mlp_W,
                               const float* __restrict__ final_W,
                               const float* __restrict__ final_b,
                               const float* __restrict__ user_bias,
                               const float* __restrict__ item_bias,
                               const float* __restrict__ label,
                               const int* __restrict__ user,
                               const int* __restrict__ item,
                               float* __restrict__ m, _Float16* __restrict__ m_h,
                               float* __restrict__ out_rating) {
  int b = blockIdx.x * blockDim.x + threadIdx.x;
  if (b >= B_) return;
  const int u = user[b], it = item[b];
  const float* ue = user_emb + (size_t)u * MF_;
  const float* ie = item_emb + (size_t)it * MF_;
  float pred = final_b[0] + AVG_ + user_bias[u] + item_bias[it];
  for (int j = 0; j < MLPO_; ++j) {
    const float* wr = mlp_W + (size_t)j * (2 * MF_);
    float acc = 0.0f;
    for (int k = 0; k < MF_; ++k) acc += ue[k] * wr[k];
    for (int k = 0; k < MF_; ++k) acc += ie[k] * wr[MF_ + k];
    float mj = tanhf(acc);
    m[b * MLPO_ + j]   = mj;
    m_h[b * MLPO_ + j] = (_Float16)mj;
    pred += mj * final_W[j];
  }
  const float d = pred - label[b];
  out_rating[b] = d * d;
}

// X_h[t*B+b, :] = [ word_emb[review_input[t,b]], m[b] ]  (f16)
__global__ void build_x_kernel(const float* __restrict__ word_emb,
                               const int* __restrict__ review_input,
                               const _Float16* __restrict__ m_h,
                               _Float16* __restrict__ X_h) {
  int i = blockIdx.x * blockDim.x + threadIdx.x;
  if (i >= T_ * B_ * XD_) return;
  const int row = i / XD_, c = i % XD_;
  const int t = row / B_, b = row % B_;
  _Float16 v;
  if (c < WD_) v = (_Float16)word_emb[(size_t)review_input[t * B_ + b] * WD_ + c];
  else         v = m_h[b * MLPO_ + (c - WD_)];
  X_h[i] = v;
}

// GRU gate update (gi/gh already carry their biases from the GEMMs)
__global__ void gru_gate_kernel(const float* __restrict__ giT,   // [B, 3H] this step
                                const float* __restrict__ gh,    // [B, 3H]
                                float* __restrict__ h, _Float16* __restrict__ h_h) {
  int i = blockIdx.x * blockDim.x + threadIdx.x;
  if (i >= B_ * H_) return;
  const int b = i / H_, j = i % H_;
  const size_t base = (size_t)b * GI_;
  const float ir = giT[base + j],           hr = gh[base + j];
  const float iz = giT[base + H_ + j],      hz = gh[base + H_ + j];
  const float in_ = giT[base + 2 * H_ + j], hn = gh[base + 2 * H_ + j];
  const float r = 1.0f / (1.0f + expf(-(ir + hr)));
  const float z = 1.0f / (1.0f + expf(-(iz + hz)));
  const float n = tanhf(in_ + r * hn);
  const float hnew = (1.0f - z) * n + z * h[i];
  h[i]   = hnew;
  h_h[i] = (_Float16)hnew;
}

// Fused log-softmax NLL per row: one block per batch row, two passes over V.
__global__ void nll_kernel(const float* __restrict__ logits,      // [B, V]
                           const int* __restrict__ tgt_all, int t,
                           float* __restrict__ out_review) {
  __shared__ float red[256];
  const int b = blockIdx.x, tid = threadIdx.x;
  const float* row = logits + (size_t)b * V_;
  float mx = -3.4e38f;
  for (int j = tid; j < V_; j += 256) mx = fmaxf(mx, row[j]);
  red[tid] = mx; __syncthreads();
  for (int s = 128; s > 0; s >>= 1) {
    if (tid < s) red[tid] = fmaxf(red[tid], red[tid + s]);
    __syncthreads();
  }
  mx = red[0]; __syncthreads();
  float sm = 0.0f;
  for (int j = tid; j < V_; j += 256) sm += expf(row[j] - mx);
  red[tid] = sm; __syncthreads();
  for (int s = 128; s > 0; s >>= 1) {
    if (tid < s) red[tid] += red[tid + s];
    __syncthreads();
  }
  if (tid == 0) {
    const int tgt = tgt_all[t * B_ + b];
    const float nll = mx + logf(red[0]) - row[tgt];
    out_review[t * B_ + b] = (tgt != 0) ? nll : 0.0f;
  }
}

__global__ void finalize_kernel(const float* __restrict__ out_all,  // d_out base
                                const int* __restrict__ tgt_all,
                                float* __restrict__ obj_out) {
  __shared__ float rs_s[256], vs_s[256];
  __shared__ int   cs_s[256];
  const int tid = threadIdx.x;
  float rs = 0.0f;
  for (int i = tid; i < B_; i += 256) rs += out_all[i];
  float vs = 0.0f; int cnt = 0;
  for (int i = tid; i < T_ * B_; i += 256) {
    vs += out_all[B_ + i];
    cnt += (tgt_all[i] != 0) ? 1 : 0;
  }
  rs_s[tid] = rs; vs_s[tid] = vs; cs_s[tid] = cnt; __syncthreads();
  for (int s = 128; s > 0; s >>= 1) {
    if (tid < s) { rs_s[tid] += rs_s[tid + s]; vs_s[tid] += vs_s[tid + s]; cs_s[tid] += cs_s[tid + s]; }
    __syncthreads();
  }
  if (tid == 0)
    obj_out[0] = 1e-7f * rs_s[0] + vs_s[0] / fmaxf((float)cs_s[0], 1.0f);
}

// -------------------------------------------------------------------------
extern "C" void kernel_launch(void* const* d_in, const int* in_sizes, int n_in,
                              void* d_out, int out_size, void* d_ws, size_t ws_size,
                              hipStream_t stream) {
  const float* user_emb  = (const float*)d_in[0];
  const float* item_emb  = (const float*)d_in[1];
  const float* mlp_W     = (const float*)d_in[2];
  const float* word_emb  = (const float*)d_in[3];
  const float* W_ih      = (const float*)d_in[4];
  const float* W_hh      = (const float*)d_in[5];
  const float* b_ih      = (const float*)d_in[6];
  const float* b_hh      = (const float*)d_in[7];
  const float* out_W     = (const float*)d_in[8];
  const float* out_b     = (const float*)d_in[9];
  const float* final_W   = (const float*)d_in[10];
  const float* final_b   = (const float*)d_in[11];
  const float* user_bias = (const float*)d_in[12];
  const float* item_bias = (const float*)d_in[13];
  const float* label     = (const float*)d_in[14];
  const int*   user      = (const int*)d_in[15];
  const int*   item      = (const int*)d_in[16];
  const int*   review_in = (const int*)d_in[17];
  const int*   review_tg = (const int*)d_in[18];

  float* out_f      = (float*)d_out;
  float* out_rating = out_f;              // [128]
  float* out_review = out_f + B_;         // [5120]
  float* out_obj    = out_f + B_ + T_ * B_;

  // ---- carve workspace (256B aligned chunks) ----
  char* ws = (char*)d_ws;
  size_t off = 0;
  auto carve = [&](size_t bytes) -> void* {
    void* p = ws + off;
    off += (bytes + 255) & ~(size_t)255;
    return p;
  };
  _Float16* Wih_h  = (_Float16*)carve((size_t)GI_ * XD_ * 2);
  _Float16* Whh_h  = (_Float16*)carve((size_t)GI_ * H_ * 2);
  _Float16* Wout_h = (_Float16*)carve((size_t)V_ * H_ * 2);
  float*    m_f    = (float*)carve((size_t)B_ * MLPO_ * 4);
  _Float16* m_h    = (_Float16*)carve((size_t)B_ * MLPO_ * 2);
  _Float16* X_h    = (_Float16*)carve((size_t)T_ * B_ * XD_ * 2);
  float*    gi     = (float*)carve((size_t)T_ * B_ * GI_ * 4);
  float*    gh     = (float*)carve((size_t)B_ * GI_ * 4);
  float*    h      = (float*)carve((size_t)B_ * H_ * 4);
  _Float16* h_h    = (_Float16*)carve((size_t)B_ * H_ * 2);
  float*    lg     = (float*)carve((size_t)B_ * V_ * 4);

  // ---- weight conversions (deterministic every call) ----
  {
    int n1 = GI_ * XD_;
    cvt_f16_kernel<<<(n1 + 255) / 256, 256, 0, stream>>>(W_ih, Wih_h, n1);
    int n2 = GI_ * H_;
    cvt_f16_kernel<<<(n2 + 255) / 256, 256, 0, stream>>>(W_hh, Whh_h, n2);
    int n3 = V_ * H_;
    cvt_f16_kernel<<<(n3 + 255) / 256, 256, 0, stream>>>(out_W, Wout_h, n3);
  }

  encoder_kernel<<<1, 128, 0, stream>>>(user_emb, item_emb, mlp_W, final_W, final_b,
                                        user_bias, item_bias, label, user, item,
                                        m_f, m_h, out_rating);
  init_h_kernel<<<(B_ * H_ + 255) / 256, 256, 0, stream>>>(h, h_h);
  build_x_kernel<<<(T_ * B_ * XD_ + 255) / 256, 256, 0, stream>>>(word_emb, review_in,
                                                                  m_h, X_h);

  // gi_all = X @ W_ih^T + b_ih : M=5120, N=1536, K=576 (32x64 macro-tiles)
  {
    int waves = (T_ * B_ / 32) * (GI_ / 64);     // 160 * 24 = 3840
    gemm_wmma_kernel<2, 4><<<(waves + 3) / 4, 128, 0, stream>>>(X_h, Wih_h, b_ih, gi,
                                                                T_ * B_, GI_, XD_);
  }

  for (int t = 0; t < T_; ++t) {
    // gh = h @ W_hh^T + b_hh : M=128, N=1536, K=512 (32x64 macro-tiles)
    {
      int waves = (B_ / 32) * (GI_ / 64);        // 4 * 24 = 96
      gemm_wmma_kernel<2, 4><<<(waves + 3) / 4, 128, 0, stream>>>(h_h, Whh_h, b_hh, gh,
                                                                  B_, GI_, H_);
    }
    gru_gate_kernel<<<(B_ * H_ + 255) / 256, 256, 0, stream>>>(
        gi + (size_t)t * B_ * GI_, gh, h, h_h);
    // logits = h_new @ out_W^T + out_b : M=128, N=20000, K=512 (32x32 macro-tiles)
    {
      int waves = (B_ / 32) * (V_ / 32);         // 4 * 625 = 2500
      gemm_wmma_kernel<2, 2><<<(waves + 3) / 4, 128, 0, stream>>>(h_h, Wout_h, out_b, lg,
                                                                  B_, V_, H_);
    }
    nll_kernel<<<B_, 256, 0, stream>>>(lg, review_tg, t, out_review);
  }

  finalize_kernel<<<1, 256, 0, stream>>>(out_f, review_tg, out_obj);
}
